// DTFD_MIL_19791209300139
// MI455X (gfx1250) — matrix operand order repaired and checked by reading, not compiled, not run on hardware
//
#include <hip/hip_runtime.h>
#include <hip/hip_bf16.h>
#include <math.h>

typedef __bf16 bf16_t;
typedef __attribute__((ext_vector_type(16))) __bf16 v16bf;
typedef __attribute__((ext_vector_type(8)))  __bf16 v8bf;
typedef __attribute__((ext_vector_type(8)))  float  v8f;
typedef __attribute__((ext_vector_type(4)))  unsigned int v4u;
typedef __attribute__((ext_vector_type(8)))  int v8i_;
typedef __attribute__((ext_vector_type(4)))  int v4i_;

#define N_TOT 65536
#define F_DIM 1024
#define D_DIM 128
#define G_NUM 8
#define NG    8192      /* N_TOT / G_NUM */
#define N2    384       /* fused secondary GEMM width: 128 V | 128 U | 2 sub_w | pad */

static __device__ __forceinline__ v8f wmma_bf16(v16bf a, v16bf b, v8f c) {
  return __builtin_amdgcn_wmma_f32_16x16x32_bf16(false, a, false, b, (short)0, c, false, false);
}

// A fragment (16x32 bf16, MxK) read from row-major LDS tile with padded stride.
// Contiguous k order, same as B: lane l, elem j -> row = (l&15), K = kbase + (l>>4)*16 + j
static __device__ __forceinline__ v16bf lds_frag_a(const bf16_t* s, int row, int k, int lda) {
  const bf16_t* p = s + row * lda + k;   // k already includes (lane>>4)*16
  v8bf lo = *(const v8bf*)(p);
  v8bf hi = *(const v8bf*)(p + 8);
  v16bf r;
  #pragma unroll
  for (int i = 0; i < 8; ++i) { r[i] = lo[i]; r[8 + i] = hi[i]; }
  return r;
}

// B fragment (32x16 bf16, KxN) from global [N][K] transposed weights.
// Contiguous k order, same as A: lane l, elem j -> N = (l&15), K = kbase + (l>>4)*16 + j  (contiguous 32B)
static __device__ __forceinline__ v16bf g_frag_b(const bf16_t* wt, int n, int k) {
  return *(const v16bf*)(wt + (size_t)n * F_DIM + k);  // k already includes (lane>>4)*16
}

static __device__ __forceinline__ float sigmoidf_(float x) { return 1.0f / (1.0f + expf(-x)); }

// ---- Tensor Data Mover: 2D tile (32 rows x 128 bf16) global->LDS with padded rows ----
// D# per CDNA5 ISA.  LDS row stride = 256B + 16B pad = 272B (136 el),
// matching lds_frag_a's conflict-free layout.  pad_interval code 5 = 64 DWORDs (256B),
// pad_amount code 3 = 4 DWORDs (16B).  data_size code 1 = 2 bytes.
// This toolchain's builtin is the 6-arg clang-23 form:
//   (uint32x4 g0, int32x8 g1, int32x4, int32x4, int32x8, i32 cpol)
static __device__ __forceinline__ void tdm_load_tile_32x128(const bf16_t* gsrc, unsigned lds_byte_off) {
  unsigned long long ga = (unsigned long long)(uintptr_t)gsrc;
  v4u g0;
  g0[0] = 1u;                                            // count=1, user D#, no gather
  g0[1] = lds_byte_off;                                  // lds_addr (bytes 63:32)
  g0[2] = (unsigned)(ga & 0xffffffffu);                  // global_addr[31:0]
  g0[3] = (unsigned)((ga >> 32) & 0x01ffffffu) | (2u << 30);  // global_addr[56:32] | type=2
  v8i_ g1;
  g1[0] = (int)((1u << 16) | (1u << 20) | (5u << 22) | (3u << 25)); // data_size|pad_en|interval|amount
  g1[1] = (int)(((unsigned)F_DIM & 0xffffu) << 16);      // tensor_dim0 = 1024 (bits 79:48)
  g1[2] = (int)(((unsigned)F_DIM >> 16) & 0xffffu);      // tensor_dim0 hi | tensor_dim1 lo16 (=0)
  g1[3] = (int)((128u << 16) | ((unsigned)N_TOT >> 16)); // tile_dim0=128 | tensor_dim1 hi16
  g1[4] = 32;                                            // tile_dim1 = 32, tile_dim2 = 0
  g1[5] = F_DIM;                                         // tensor_dim0_stride lo32 = 1024
  g1[6] = 0;                                             // stride hi | dim1_stride lo
  g1[7] = 0;
  v4i_ z4; z4[0] = 0; z4[1] = 0; z4[2] = 0; z4[3] = 0;   // groups 2/3 unused (2D tensor)
  v8i_ z8;
  #pragma unroll
  for (int i = 0; i < 8; ++i) z8[i] = 0;
  __builtin_amdgcn_tensor_load_to_lds(g0, g1, z4, z4, z8, 0);
}

// ---------------- k0a: wdT[n][k] = bf16(w_dim[k][n]) (LDS tiled transpose) ----------------
__global__ __launch_bounds__(256) void k0a_transpose(const float* __restrict__ w, bf16_t* __restrict__ wdT) {
  __shared__ float tile[32][33];
  const int tx = threadIdx.x & 31, ty = threadIdx.x >> 5;  // 32 x 8
  const int bx = blockIdx.x, by = blockIdx.y;
  #pragma unroll
  for (int j = 0; j < 4; ++j)
    tile[ty + j * 8][tx] = w[(size_t)(by * 32 + ty + j * 8) * F_DIM + bx * 32 + tx];
  __syncthreads();
  #pragma unroll
  for (int j = 0; j < 4; ++j)
    wdT[(size_t)(bx * 32 + ty + j * 8) * F_DIM + by * 32 + tx] = (bf16_t)tile[tx][ty + j * 8];
}

// ---------------- k0b: fused secondary B matrix w2t[384][1024] ----------------
__global__ __launch_bounds__(256) void k0b_buildw2(const float* __restrict__ Vw, const float* __restrict__ Uw,
                                                   const float* __restrict__ subw, bf16_t* __restrict__ w2t) {
  int idx = blockIdx.x * 256 + threadIdx.x;          // 384*1024 total
  int n = idx >> 10, k = idx & 1023;
  float v = 0.0f;
  if (n < 128)            v = Vw[(size_t)k * D_DIM + n];
  else if (n < 256)       v = Uw[(size_t)k * D_DIM + (n - 128)];
  else if (n == 256)      v = subw[k * 2 + 0];
  else if (n == 257)      v = subw[k * 2 + 1];
  w2t[(size_t)n * F_DIM + k] = (bf16_t)v;
}

// ---------------- k1: mid = relu(x[perm] @ w_dim), bf16 out ----------------
// block = 256 thr = 8 waves (2 x 4), tile M=64 N=256; A staged fp32->bf16 in LDS.
__global__ __launch_bounds__(256) void k1_dimred(const float* __restrict__ x, const int* __restrict__ perm,
                                                 const bf16_t* __restrict__ wdT, bf16_t* __restrict__ mid) {
  __shared__ bf16_t As[64 * 136];   // padded stride 136 -> conflict-free frag reads
  __shared__ int rows[64];
  const int tid = threadIdx.x, lane = tid & 31, wid = tid >> 5;
  const int waveN = wid & 3, waveM = wid >> 2;
  const int l15 = lane & 15, lHi = lane >> 4;
  const int rowBase = blockIdx.y * 64;
  const int colBase = blockIdx.x * 256;
  if (tid < 64) rows[tid] = perm[rowBase + tid];

  v8f acc[2][4];
  #pragma unroll
  for (int mt = 0; mt < 2; ++mt)
    #pragma unroll
    for (int nt = 0; nt < 4; ++nt)
      #pragma unroll
      for (int i = 0; i < 8; ++i) acc[mt][nt][i] = 0.0f;

  for (int kc = 0; kc < F_DIM; kc += 128) {
    __syncthreads();
    #pragma unroll 4
    for (int i = 0; i < 32; ++i) {        // stage 64x128 fp32 -> bf16
      int e = tid + 256 * i;
      int r = e >> 7, k = e & 127;
      As[r * 136 + k] = (bf16_t)x[(size_t)rows[r] * F_DIM + kc + k];
    }
    if (tid < 64 && kc + 128 < F_DIM)
      __builtin_prefetch(&x[(size_t)rows[tid] * F_DIM + kc + 128], 0, 1);
    __syncthreads();
    #pragma unroll
    for (int ks = 0; ks < 4; ++ks) {
      const int kk = ks * 32;
      v16bf a0 = lds_frag_a(As, 32 * waveM + l15,      kk + lHi * 16, 136);
      v16bf a1 = lds_frag_a(As, 32 * waveM + 16 + l15, kk + lHi * 16, 136);
      v16bf b[4];
      #pragma unroll
      for (int nt = 0; nt < 4; ++nt) {    // issue all B loads first (one clause, hide L2 latency)
        int n = colBase + 64 * waveN + 16 * nt + l15;
        b[nt] = g_frag_b(wdT, n, kc + kk + lHi * 16);
      }
      #pragma unroll
      for (int nt = 0; nt < 4; ++nt) {
        acc[0][nt] = wmma_bf16(a0, b[nt], acc[0][nt]);
        acc[1][nt] = wmma_bf16(a1, b[nt], acc[1][nt]);
      }
    }
  }
  // epilogue: ReLU + bf16 store.  C layout: elem r -> M = r + 8*(lane>>4), N = lane&15
  #pragma unroll
  for (int mt = 0; mt < 2; ++mt)
    #pragma unroll
    for (int nt = 0; nt < 4; ++nt) {
      int c = colBase + 64 * waveN + 16 * nt + l15;
      #pragma unroll
      for (int r = 0; r < 8; ++r) {
        int p = rowBase + 32 * waveM + 16 * mt + r + 8 * lHi;
        mid[(size_t)p * F_DIM + c] = (bf16_t)fmaxf(acc[mt][nt][r], 0.0f);
      }
    }
}

// ---------------- k2: fused gated-attention logits + CAM key precursor ----------------
// [32 x 1024] @ [1024 x 384] WMMA with TDM double-buffered A staging;
// epilogue computes a_i and d_i = c1 - c0.
__global__ __launch_bounds__(256) void k2_att(const bf16_t* __restrict__ mid, const bf16_t* __restrict__ w2t,
                                              const float* __restrict__ vB, const float* __restrict__ uB,
                                              const float* __restrict__ attw, const float* __restrict__ attwb,
                                              float* __restrict__ aArr, float* __restrict__ dArr) {
  __shared__ float smem[32 * 392];            // union: 2x TDM A-buffers (bf16) then hvals (fp32)
  const int tid = threadIdx.x, lane = tid & 31, wid = tid >> 5;   // 8 waves, 3 N-tiles each
  const int l15 = lane & 15, lHi = lane >> 4;
  const int p0 = blockIdx.x * 32;
  const bf16_t* srcBase = mid + (size_t)p0 * F_DIM;

  v8f acc[2][3];
  #pragma unroll
  for (int mt = 0; mt < 2; ++mt)
    #pragma unroll
    for (int nt = 0; nt < 3; ++nt)
      #pragma unroll
      for (int i = 0; i < 8; ++i) acc[mt][nt][i] = 0.0f;

  if (tid < 32) tdm_load_tile_32x128(srcBase, 0);             // prime buffer 0
  for (int c = 0; c < 8; ++c) {
    if (tid < 32) {
      if (c + 1 < 8) {                                         // issue next, wait current
        tdm_load_tile_32x128(srcBase + (c + 1) * 128, (unsigned)(((c + 1) & 1) * 8704));
        __builtin_amdgcn_s_wait_tensorcnt(1);
      } else {
        __builtin_amdgcn_s_wait_tensorcnt(0);
      }
    }
    __syncthreads();                                           // publish TDM'd tile to all waves
    const bf16_t* As = (const bf16_t*)smem + (c & 1) * 4352;
    const int kc = c * 128;
    #pragma unroll
    for (int ks = 0; ks < 4; ++ks) {
      const int kk = ks * 32;
      v16bf a0 = lds_frag_a(As, l15,      kk + lHi * 16, 136);
      v16bf a1 = lds_frag_a(As, 16 + l15, kk + lHi * 16, 136);
      v16bf b[3];
      #pragma unroll
      for (int nt = 0; nt < 3; ++nt) {
        int n = 48 * wid + 16 * nt + l15;
        b[nt] = g_frag_b(w2t, n, kc + kk + lHi * 16);
      }
      #pragma unroll
      for (int nt = 0; nt < 3; ++nt) {
        acc[0][nt] = wmma_bf16(a0, b[nt], acc[0][nt]);
        acc[1][nt] = wmma_bf16(a1, b[nt], acc[1][nt]);
      }
    }
    __syncthreads();                                           // all reads done before buffer reuse
  }
  #pragma unroll
  for (int mt = 0; mt < 2; ++mt)
    #pragma unroll
    for (int nt = 0; nt < 3; ++nt)
      #pragma unroll
      for (int r = 0; r < 8; ++r)
        smem[(16 * mt + r + 8 * lHi) * 392 + 48 * wid + 16 * nt + l15] = acc[mt][nt][r];
  __syncthreads();
  {
    int r = tid >> 3, jl = tid & 7;
    float s = 0.0f;
    for (int j = jl; j < D_DIM; j += 8) {
      float hv = smem[r * 392 + j] + vB[j];
      float hu = smem[r * 392 + D_DIM + j] + uB[j];
      s += tanhf(hv) * sigmoidf_(hu) * attw[j];
    }
    s += __shfl_xor(s, 1); s += __shfl_xor(s, 2); s += __shfl_xor(s, 4);
    if (jl == 0) {
      aArr[p0 + r] = s + attwb[0];
      dArr[p0 + r] = smem[r * 392 + 257] - smem[r * 392 + 256];  // c1 - c0
    }
  }
}

// ---------------- k3: per-group softmax -> A_i; key_i = A_i * d_i (in place) ----------------
__global__ __launch_bounds__(256) void k3_softmax(const float* __restrict__ aArr, float* __restrict__ dArr,
                                                  float* __restrict__ Aw) {
  __shared__ float red[256];
  const int g = blockIdx.x, tid = threadIdx.x;
  const int base = g * NG;
  float m = -__builtin_inff();
  for (int i = tid; i < NG; i += 256) m = fmaxf(m, aArr[base + i]);
  red[tid] = m; __syncthreads();
  for (int s = 128; s > 0; s >>= 1) { if (tid < s) red[tid] = fmaxf(red[tid], red[tid + s]); __syncthreads(); }
  const float mx = red[0]; __syncthreads();
  float se = 0.0f;
  for (int i = tid; i < NG; i += 256) se += expf(aArr[base + i] - mx);
  red[tid] = se; __syncthreads();
  for (int s = 128; s > 0; s >>= 1) { if (tid < s) red[tid] += red[tid + s]; __syncthreads(); }
  const float inv = 1.0f / red[0];
  for (int i = tid; i < NG; i += 256) {
    float A = expf(aArr[base + i] - mx) * inv;
    Aw[base + i] = A;
    dArr[base + i] = A * dArr[base + i];     // sort key, monotone in probs[:,1]
  }
}

// ---------------- k4: att_feat_g = sum_i A_i * mid_i (column striped) ----------------
__global__ __launch_bounds__(256) void k4_attfeat(const bf16_t* __restrict__ mid, const float* __restrict__ Aw,
                                                  float* __restrict__ attF) {
  const int g = blockIdx.y;
  const int col = blockIdx.x * 256 + threadIdx.x;
  const size_t base = (size_t)g * NG;
  float acc = 0.0f;
  #pragma unroll 8
  for (int r = 0; r < NG; ++r) {
    size_t p = base + r;
    acc += Aw[p] * (float)mid[p * F_DIM + col];
  }
  attF[g * F_DIM + col] = acc;
}

// ---------------- k4b: sub_preds = att_feat @ sub_w + sub_b ----------------
__global__ __launch_bounds__(64) void k4b_subpred(const float* __restrict__ attF, const float* __restrict__ subw,
                                                  const float* __restrict__ subb, float* __restrict__ out) {
  __shared__ float r0[64], r1[64];
  const int g = blockIdx.x, tid = threadIdx.x;
  float s0 = 0.0f, s1 = 0.0f;
  for (int c = tid; c < F_DIM; c += 64) {
    float v = attF[g * F_DIM + c];
    s0 += v * subw[c * 2 + 0];
    s1 += v * subw[c * 2 + 1];
  }
  r0[tid] = s0; r1[tid] = s1; __syncthreads();
  for (int s = 32; s > 0; s >>= 1) { if (tid < s) { r0[tid] += r0[tid + s]; r1[tid] += r1[tid + s]; } __syncthreads(); }
  if (tid == 0) {
    out[1026 + g * 2 + 0] = r0[0] + subb[0];
    out[1026 + g * 2 + 1] = r1[0] + subb[1];
  }
}

// ---------------- k5: per-group top-8 (max) + bottom-8 (min), argsort-compatible ties ----------------
__global__ __launch_bounds__(256) void k5_select(const float* __restrict__ key, int* __restrict__ pidx) {
  __shared__ float sv[256];
  __shared__ int   si[256];
  const int g = blockIdx.x, tid = threadIdx.x;
  const int base = g * NG;
  float kv[32];
  #pragma unroll
  for (int i = 0; i < 32; ++i) kv[i] = key[base + tid + 256 * i];

  // top-8: max, tie -> smaller index; slots 0..7
  for (int pass = 0; pass < 8; ++pass) {
    float bv = kv[0]; int bp = tid;
    #pragma unroll
    for (int i = 1; i < 32; ++i) {
      float v = kv[i]; int p = tid + 256 * i;
      if (v > bv || (v == bv && p < bp)) { bv = v; bp = p; }
    }
    sv[tid] = bv; si[tid] = bp; __syncthreads();
    for (int s = 128; s > 0; s >>= 1) {
      if (tid < s) {
        float v = sv[tid + s]; int p = si[tid + s];
        if (v > sv[tid] || (v == sv[tid] && p < si[tid])) { sv[tid] = v; si[tid] = p; }
      }
      __syncthreads();
    }
    int wbp = si[0]; __syncthreads();
    if (tid == 0) pidx[g * 16 + pass] = base + wbp;
    if ((wbp & 255) == tid) kv[wbp >> 8] = -__builtin_inff();
  }
  // restore, then bottom-8: min, tie -> larger index; slots 15..8 (smallest last)
  #pragma unroll
  for (int i = 0; i < 32; ++i) kv[i] = key[base + tid + 256 * i];
  for (int pass = 0; pass < 8; ++pass) {
    float bv = kv[0]; int bp = tid;
    #pragma unroll
    for (int i = 1; i < 32; ++i) {
      float v = kv[i]; int p = tid + 256 * i;
      if (v < bv || (v == bv && p > bp)) { bv = v; bp = p; }
    }
    sv[tid] = bv; si[tid] = bp; __syncthreads();
    for (int s = 128; s > 0; s >>= 1) {
      if (tid < s) {
        float v = sv[tid + s]; int p = si[tid + s];
        if (v < sv[tid] || (v == sv[tid] && p > si[tid])) { sv[tid] = v; si[tid] = p; }
      }
      __syncthreads();
    }
    int wbp = si[0]; __syncthreads();
    if (tid == 0) pidx[g * 16 + 15 - pass] = base + wbp;
    if ((wbp & 255) == tid) kv[wbp >> 8] = __builtin_inff();
  }
}

// ---------------- k6a: second-tier gated-attention logits over 128 distilled rows ----------------
__global__ __launch_bounds__(128) void k6a_att2(const bf16_t* __restrict__ mid, const int* __restrict__ pidx,
                                                const float* __restrict__ Vw2, const float* __restrict__ Uw2,
                                                const float* __restrict__ vB2, const float* __restrict__ uB2,
                                                const float* __restrict__ aw2, const float* __restrict__ aw2b,
                                                float* __restrict__ a2) {
  __shared__ float ms[128];
  __shared__ float red[128];
  const int tid = threadIdx.x;
  const int prow = pidx[blockIdx.x];
  float hv = 0.0f, hu = 0.0f;
  for (int kt = 0; kt < 8; ++kt) {
    __syncthreads();
    ms[tid] = (float)mid[(size_t)prow * F_DIM + kt * 128 + tid];
    __syncthreads();
    #pragma unroll 4
    for (int j = 0; j < 128; ++j) {
      float m = ms[j];
      int k = kt * 128 + j;
      hv += m * Vw2[(size_t)k * D_DIM + tid];
      hu += m * Uw2[(size_t)k * D_DIM + tid];
    }
  }
  float gg = tanhf(hv + vB2[tid]) * sigmoidf_(hu + uB2[tid]);
  red[tid] = gg * aw2[tid]; __syncthreads();
  for (int s = 64; s > 0; s >>= 1) { if (tid < s) red[tid] += red[tid + s]; __syncthreads(); }
  if (tid == 0) a2[blockIdx.x] = red[0] + aw2b[0];
}

// ---------------- k6b: softmax(a2) -> AA; afeat = AA @ pseudo; bag_pred ----------------
__global__ __launch_bounds__(256) void k6b_bag(const bf16_t* __restrict__ mid, const int* __restrict__ pidx,
                                               const float* __restrict__ a2, const float* __restrict__ clsw,
                                               const float* __restrict__ clsb, float* __restrict__ out) {
  __shared__ float AA[128];
  __shared__ int   pr[128];
  __shared__ float red[256];
  __shared__ float invZ;
  const int tid = threadIdx.x;
  if (tid < 128) { AA[tid] = a2[tid]; pr[tid] = pidx[tid]; }
  __syncthreads();
  if (tid == 0) {
    float mx = -__builtin_inff();
    for (int s = 0; s < 128; ++s) mx = fmaxf(mx, AA[s]);
    float se = 0.0f;
    for (int s = 0; s < 128; ++s) { float e = expf(AA[s] - mx); AA[s] = e; se += e; }
    invZ = 1.0f / se;
  }
  __syncthreads();
  if (tid < 128) AA[tid] *= invZ;
  __syncthreads();
  float pb0 = 0.0f, pb1 = 0.0f;
  for (int c = tid; c < F_DIM; c += 256) {
    float acc = 0.0f;
    #pragma unroll 4
    for (int s = 0; s < 128; ++s) acc += AA[s] * (float)mid[(size_t)pr[s] * F_DIM + c];
    out[2 + c] = acc;                 // afeat
    pb0 += acc * clsw[c * 2 + 0];
    pb1 += acc * clsw[c * 2 + 1];
  }
  red[tid] = pb0; __syncthreads();
  for (int s = 128; s > 0; s >>= 1) { if (tid < s) red[tid] += red[tid + s]; __syncthreads(); }
  float t0 = red[0]; __syncthreads();
  red[tid] = pb1; __syncthreads();
  for (int s = 128; s > 0; s >>= 1) { if (tid < s) red[tid] += red[tid + s]; __syncthreads(); }
  if (tid == 0) { out[0] = t0 + clsb[0]; out[1] = red[0] + clsb[1]; }
}

extern "C" void kernel_launch(void* const* d_in, const int* in_sizes, int n_in,
                              void* d_out, int out_size, void* d_ws, size_t ws_size,
                              hipStream_t stream) {
  const float* x      = (const float*)d_in[0];
  const int*   perm   = (const int*)  d_in[1];
  const float* w_dim  = (const float*)d_in[2];
  const float* attVw  = (const float*)d_in[3];
  const float* attVb  = (const float*)d_in[4];
  const float* attUw  = (const float*)d_in[5];
  const float* attUb  = (const float*)d_in[6];
  const float* attw   = (const float*)d_in[7];
  const float* attwb  = (const float*)d_in[8];
  const float* subw   = (const float*)d_in[9];
  const float* subb   = (const float*)d_in[10];
  const float* at2Vw  = (const float*)d_in[11];
  const float* at2Vb  = (const float*)d_in[12];
  const float* at2Uw  = (const float*)d_in[13];
  const float* at2Ub  = (const float*)d_in[14];
  const float* at2w   = (const float*)d_in[15];
  const float* at2wb  = (const float*)d_in[16];
  const float* clsw   = (const float*)d_in[17];
  const float* clsb   = (const float*)d_in[18];
  float* out = (float*)d_out;

  char* ws = (char*)d_ws;
  bf16_t* mid  = (bf16_t*)(ws + 0);           // 65536*1024*2 = 134217728
  bf16_t* wdT  = (bf16_t*)(ws + 134217728);   // 1024*1024*2  = 2097152
  bf16_t* w2t  = (bf16_t*)(ws + 136314880);   // 384*1024*2   = 786432
  float*  aArr = (float*) (ws + 137101312);   // 65536*4
  float*  dArr = (float*) (ws + 137363456);   // 65536*4
  float*  Aw   = (float*) (ws + 137625600);   // 65536*4
  float*  attF = (float*) (ws + 137887744);   // 8*1024*4
  float*  a2   = (float*) (ws + 137920512);   // 128*4
  int*    pidx = (int*)   (ws + 137921024);   // 128*4

  k0a_transpose<<<dim3(32, 32), 256, 0, stream>>>(w_dim, wdT);
  k0b_buildw2  <<<(N2 * F_DIM) / 256, 256, 0, stream>>>(attVw, attUw, subw, w2t);
  k1_dimred    <<<dim3(4, N_TOT / 64), 256, 0, stream>>>(x, perm, wdT, mid);
  k2_att       <<<N_TOT / 32, 256, 0, stream>>>(mid, w2t, attVb, attUb, attw, attwb, aArr, dArr);
  k3_softmax   <<<G_NUM, 256, 0, stream>>>(aArr, dArr, Aw);
  k4_attfeat   <<<dim3(4, G_NUM), 256, 0, stream>>>(mid, Aw, attF);
  k4b_subpred  <<<G_NUM, 64, 0, stream>>>(attF, subw, subb, out);
  k5_select    <<<G_NUM, 256, 0, stream>>>(dArr, pidx);
  k6a_att2     <<<128, 128, 0, stream>>>(mid, pidx, at2Vw, at2Uw, at2Vb, at2Ub, at2w, at2wb, a2);
  k6b_bag      <<<1, 256, 0, stream>>>(mid, pidx, a2, clsw, clsb, out);
}
